// NewAdvKPCN_1_15109694947676
// MI455X (gfx1250) — matrix operand order repaired and checked by reading, not compiled, not run on hardware
//
#include <hip/hip_runtime.h>
#include <hip/hip_bf16.h>

// ---------------------------------------------------------------------------
// Types for WMMA (CDNA5 / gfx1250, wave32)
// ---------------------------------------------------------------------------
typedef _Float16 v16h __attribute__((ext_vector_type(16)));
typedef _Float16 v8h  __attribute__((ext_vector_type(8)));
typedef float    v8f  __attribute__((ext_vector_type(8)));
typedef int      v4i  __attribute__((ext_vector_type(4)));

#define BLK_M 128
#define BLK_N 64
#define BLK_K 64

// Probe for gfx1250 async global->LDS builtins (guarded: absent -> register path)
#if defined(__has_builtin)
#  if __has_builtin(__builtin_amdgcn_global_load_async_to_lds_b128) && \
      __has_builtin(__builtin_amdgcn_s_wait_asynccnt)
#    define HAVE_ASYNC_LDS 1
#  endif
#endif
#ifndef HAVE_ASYNC_LDS
#  define HAVE_ASYNC_LDS 0
#endif

static inline int cdiv(int a, int b) { return (a + b - 1) / b; }

// ---------------------------------------------------------------------------
// Implicit-GEMM convolution (NCHW), f16 in, f32 accumulate via WMMA.
// M = Hout*Wout, N = Cout, K = Cin*ks*ks.
// X:  f16 activations [Cin, Hin, Win]
// Wt: f16 weights, packed [Npad][Kpad] (n-major, k contiguous, zero padded)
// Y16: f16 out [Cout, Hout, Wout]
// Block: 256 threads = 8 waves; tile 128(M) x 64(N), K-chunks of 64,
// software-pipelined double-buffered LDS, 8 WMMAs per chunk per wave.
// B-panel staged with CDNA5 async global->LDS (ASYNCcnt) when available.
// ---------------------------------------------------------------------------
__global__ __launch_bounds__(256)
void conv_wmma_f16(const _Float16* __restrict__ X,
                   const _Float16* __restrict__ Wt,
                   const float* __restrict__ bias,
                   _Float16* __restrict__ Y16,
                   int Cin, int Hin, int Win,
                   int Cout, int Hout, int Wout,
                   int ks, int stride, int pad, int relu, int Kpad)
{
  __shared__ __align__(128) _Float16 As[2][BLK_M * BLK_K];  // [m][k]
  __shared__ __align__(128) _Float16 Bt[2][BLK_N * BLK_K];  // [n][k]
  __shared__ int s_oy[BLK_M], s_ox[BLK_M];

  const int M   = Hout * Wout;
  const int N   = Cout;
  const int K   = Cin * ks * ks;
  const int ksq = ks * ks;

  const int m0   = blockIdx.x * BLK_M;
  const int n0   = blockIdx.y * BLK_N;
  const int tid  = threadIdx.x;
  const int lane = tid & 31;
  const int wave = tid >> 5;
  const int wm   = (wave & 3) << 5;   // wave M offset: 0,32,64,96
  const int wn   = (wave >> 2) << 5;  // wave N offset: 0,32

  // Fixed staging coordinates per thread
  const int aki = tid & 63;          // A: k within chunk (FIXED -> 1 div/mod per chunk)
  const int am  = tid >> 6;          // A: row = it*4 + am
  const int bni = tid >> 2;          // B: column (0..63)
  const int bko = (tid & 3) * 16;    // B: k offset within chunk

  // per-block output coordinate table (scaled by stride, pre-shifted by pad)
  for (int i = tid; i < BLK_M; i += 256) {
    int m = m0 + i;
    int mc = (m < M) ? m : 0;
    s_oy[i] = (mc / Wout) * stride - pad;
    s_ox[i] = (mc % Wout) * stride - pad;
  }

  v8f acc00 = {}, acc01 = {}, acc10 = {}, acc11 = {};
  _Float16 areg[32];
#if !HAVE_ASYNC_LDS
  v8h breg0, breg1;
#endif

  // ---- stage chunk kc into registers (A) ----
  auto stageA = [&](int kc) {
    int kk = kc + aki;
    bool vk = kk < K;
    int ci = 0, kh = 0, kw = 0;
    if (vk) {
      ci = kk / ksq;
      int rem = kk - ci * ksq;
      kh = rem / ks;
      kw = rem - kh * ks;
    }
    size_t cbase = (size_t)ci * Hin * Win;
#pragma unroll
    for (int it = 0; it < 32; ++it) {
      int mi = it * 4 + am;
      _Float16 v = (_Float16)0.f;
      if (vk && (m0 + mi) < M) {
        int iy = s_oy[mi] + kh;
        int ix = s_ox[mi] + kw;
        if ((unsigned)iy < (unsigned)Hin && (unsigned)ix < (unsigned)Win)
          v = X[cbase + (size_t)iy * Win + ix];
      }
      areg[it] = v;
    }
  };
  auto writeA = [&](int buf) {
#pragma unroll
    for (int it = 0; it < 32; ++it)
      As[buf][(it * 4 + am) * BLK_K + aki] = areg[it];
  };
#if HAVE_ASYNC_LDS
  auto stageB_async = [&](int kc, int buf) {
    __attribute__((address_space(1))) v4i* g =
        (__attribute__((address_space(1))) v4i*)(Wt + (size_t)(n0 + bni) * Kpad + kc + bko);
    __attribute__((address_space(3))) v4i* l =
        (__attribute__((address_space(3))) v4i*)&Bt[buf][bni * BLK_K + bko];
    __builtin_amdgcn_global_load_async_to_lds_b128(g, l, 0, 0);
    __builtin_amdgcn_global_load_async_to_lds_b128(g + 1, l + 1, 0, 0);
  };
#else
  auto stageB = [&](int kc) {
    const v8h* g = (const v8h*)(Wt + (size_t)(n0 + bni) * Kpad + kc + bko);
    breg0 = g[0];
    breg1 = g[1];
  };
  auto writeB = [&](int buf) {
    *(v8h*)&Bt[buf][bni * BLK_K + bko]     = breg0;
    *(v8h*)&Bt[buf][bni * BLK_K + bko + 8] = breg1;
  };
#endif

  auto compute = [&](int buf) {
    const int arow = lane & 15;
    const int akb  = (lane < 16) ? 0 : 16;   // A: 16 contiguous K per lane
    const int bkb  = (lane < 16) ? 0 : 16;   // B: 16 contiguous K per lane
#pragma unroll
    for (int s = 0; s < 2; ++s) {
      int ko = s * 32;
      v8h a0lo = *(const v8h*)&As[buf][(wm + arow + 0)  * BLK_K + ko + akb];
      v8h a0hi = *(const v8h*)&As[buf][(wm + arow + 0)  * BLK_K + ko + akb + 8];
      v8h a1lo = *(const v8h*)&As[buf][(wm + arow + 16) * BLK_K + ko + akb];
      v8h a1hi = *(const v8h*)&As[buf][(wm + arow + 16) * BLK_K + ko + akb + 8];
      v16h a0 = __builtin_shufflevector(a0lo, a0hi, 0,1,2,3,4,5,6,7,8,9,10,11,12,13,14,15);
      v16h a1 = __builtin_shufflevector(a1lo, a1hi, 0,1,2,3,4,5,6,7,8,9,10,11,12,13,14,15);
      v16h b0 = *(const v16h*)&Bt[buf][(wn + arow + 0)  * BLK_K + ko + bkb];
      v16h b1 = *(const v16h*)&Bt[buf][(wn + arow + 16) * BLK_K + ko + bkb];
      acc00 = __builtin_amdgcn_wmma_f32_16x16x32_f16(false, a0, false, b0, (short)0, acc00, false, false);
      acc01 = __builtin_amdgcn_wmma_f32_16x16x32_f16(false, a0, false, b1, (short)0, acc01, false, false);
      acc10 = __builtin_amdgcn_wmma_f32_16x16x32_f16(false, a1, false, b0, (short)0, acc10, false, false);
      acc11 = __builtin_amdgcn_wmma_f32_16x16x32_f16(false, a1, false, b1, (short)0, acc11, false, false);
    }
  };

  // ---- prologue: table + chunk 0 ----
  __syncthreads();               // publish s_oy/s_ox
  stageA(0);
  writeA(0);
#if HAVE_ASYNC_LDS
  stageB_async(0, 0);
  __builtin_amdgcn_s_wait_asynccnt(0);
#else
  stageB(0);
  writeB(0);
#endif
  __syncthreads();

  // ---- main pipelined loop ----
  int buf = 0;
  for (int kc = 0; kc < K; kc += BLK_K) {
    bool more = (kc + BLK_K) < K;
    if (more) {
      stageA(kc + BLK_K);
#if HAVE_ASYNC_LDS
      stageB_async(kc + BLK_K, buf ^ 1);
#else
      __builtin_prefetch(&Wt[(size_t)(n0 + bni) * Kpad + kc + BLK_K], 0, 1);
      stageB(kc + BLK_K);
#endif
    }
    compute(buf);
    if (more) {
      writeA(buf ^ 1);
#if HAVE_ASYNC_LDS
      __builtin_amdgcn_s_wait_asynccnt(0);
#else
      writeB(buf ^ 1);
#endif
    }
    __syncthreads();
    buf ^= 1;
  }

  // ---- epilogue ----
  const int cl    = lane & 15;
  const int rsel  = (lane < 16) ? 0 : 8;
  const bool fast = (m0 + BLK_M <= M) && ((M & 7) == 0);
#pragma unroll
  for (int ms = 0; ms < 2; ++ms) {
    int mbase = m0 + wm + ms * 16 + rsel;
#pragma unroll
    for (int ns = 0; ns < 2; ++ns) {
      int col = n0 + wn + ns * 16 + cl;
      v8f c = (ms == 0) ? (ns == 0 ? acc00 : acc01)
                        : (ns == 0 ? acc10 : acc11);
      if (col < N) {
        float bb = bias ? bias[col] : 0.0f;
        size_t base = (size_t)col * M;
        if (fast) {
          v8h o;
#pragma unroll
          for (int r = 0; r < 8; ++r) {
            float v = c[r] + bb;
            if (relu) v = fmaxf(v, 0.0f);
            o[r] = (_Float16)v;
          }
          *(v8h*)&Y16[base + mbase] = o;
        } else {
#pragma unroll
          for (int r = 0; r < 8; ++r) {
            int m = mbase + r;
            if (m < M) {
              float v = c[r] + bb;
              if (relu) v = fmaxf(v, 0.0f);
              Y16[base + m] = (_Float16)v;
            }
          }
        }
      }
    }
  }
}

// ---------------------------------------------------------------------------
// Pack OIHW f32 weights -> f16 [Npad][Kpad] (n-major, k contiguous, zero pad).
// k = (ci*ks+kh)*ks+kw matches OIHW flattening, so this is a strided cast.
// ---------------------------------------------------------------------------
__global__ void pack_weights(const float* __restrict__ w, _Float16* __restrict__ dst,
                             int Cout, int CinKK, int Kpad, int total)
{
  int i = blockIdx.x * blockDim.x + threadIdx.x;
  if (i >= total) return;
  int n = i / Kpad;
  int k = i - n * Kpad;
  float v = 0.f;
  if (n < Cout && k < CinKK) v = w[(size_t)n * CinKK + k];
  dst[i] = (_Float16)v;
}

// cast f32 -> f16 with optional spatial crop (src [C,srcH,srcW] -> dst [C,outH,outW])
__global__ void cast_crop(const float* __restrict__ src, _Float16* __restrict__ dst,
                          int C, int srcH, int srcW, int off, int outH, int outW)
{
  int i = blockIdx.x * blockDim.x + threadIdx.x;
  int total = C * outH * outW;
  if (i >= total) return;
  int c = i / (outH * outW);
  int r = i - c * (outH * outW);
  int y = r / outW, x = r - y * outW;
  dst[i] = (_Float16)src[((size_t)c * srcH + y + off) * srcW + x + off];
}

// per-pixel softmax over C channels (channel-major [C, M])
__global__ void softmaxC(const _Float16* __restrict__ in, _Float16* __restrict__ out,
                         int C, int M)
{
  int p = blockIdx.x * blockDim.x + threadIdx.x;
  if (p >= M) return;
  float mx = -3.0e38f;
  for (int c = 0; c < C; ++c) mx = fmaxf(mx, (float)in[(size_t)c * M + p]);
  float s = 0.f;
  for (int c = 0; c < C; ++c) {
    float e = __expf((float)in[(size_t)c * M + p] - mx);
    s += e;
    out[(size_t)c * M + p] = (_Float16)e;
  }
  float inv = 1.f / s;
  for (int c = 0; c < C; ++c)
    out[(size_t)c * M + p] = (_Float16)((float)out[(size_t)c * M + p] * inv);
}

// KPCN weighted filter: buf f32 [3,128,128] cropped@18 to 92^2, pad r=10,
// w f16 [441, 92*92] -> out f32 [3, 92*92]
__global__ void wfilter(const float* __restrict__ buf, const _Float16* __restrict__ w,
                        float* __restrict__ out)
{
  const int M = 92 * 92;
  int p = blockIdx.x * blockDim.x + threadIdx.x;
  if (p >= M) return;
  int y = p / 92, x = p - y * 92;
  float a0 = 0.f, a1 = 0.f, a2 = 0.f;
  for (int k = 0; k < 441; ++k) {
    int ky = k / 21 - 10;
    int kx = (k % 21) - 10;
    int yy = y + ky, xx = x + kx;
    if ((unsigned)yy < 92u && (unsigned)xx < 92u) {
      float wv = (float)w[(size_t)k * M + p];
      int base = (18 + yy) * 128 + (18 + xx);
      a0 += wv * buf[base];
      a1 += wv * buf[128 * 128 + base];
      a2 += wv * buf[2 * 128 * 128 + base];
    }
  }
  out[p] = a0; out[M + p] = a1; out[2 * M + p] = a2;
}

// nearest 2x upsample, f16 [C,Hin,Win] -> [C,2Hin,2Win]
__global__ void up2(const _Float16* __restrict__ in, _Float16* __restrict__ out,
                    int C, int Hin, int Win)
{
  int Ho = Hin * 2, Wo = Win * 2;
  int i = blockIdx.x * blockDim.x + threadIdx.x;
  int total = C * Ho * Wo;
  if (i >= total) return;
  int c = i / (Ho * Wo);
  int r = i - c * (Ho * Wo);
  int y = r / Wo, x = r - y * Wo;
  out[i] = in[((size_t)c * Hin + (y >> 1)) * Win + (x >> 1)];
}

// dst[c][p] = sigmoid(in51[1+c][p]) for c in [0,C)
__global__ void sigmoid_slice(const _Float16* __restrict__ in51, _Float16* __restrict__ dst,
                              int C, int M)
{
  int i = blockIdx.x * blockDim.x + threadIdx.x;
  if (i >= C * M) return;
  int c = i / M, p = i - c * M;
  float v = (float)in51[(size_t)(1 + c) * M + p];
  dst[i] = (_Float16)(1.f / (1.f + __expf(-v)));
}

// n_k = g*sigmoid(wraw) + p*(1-sigmoid(wraw)); wraw is [1, M]
__global__ void blendk(const _Float16* __restrict__ g, const _Float16* __restrict__ pk,
                       const _Float16* __restrict__ wraw, _Float16* __restrict__ out,
                       int C, int M)
{
  int i = blockIdx.x * blockDim.x + threadIdx.x;
  if (i >= C * M) return;
  int p = i % M;
  float w = 1.f / (1.f + __expf(-(float)wraw[p]));
  out[i] = (_Float16)((float)g[i] * w + (float)pk[i] * (1.f - w));
}

// out = alb_crop * r_d + exp(r_s) - 1, all f32, out [3, 92*92]
__global__ void final_out(const float* __restrict__ alb, const float* __restrict__ rd,
                          const float* __restrict__ rs, float* __restrict__ out)
{
  const int M = 92 * 92;
  int i = blockIdx.x * blockDim.x + threadIdx.x;
  if (i >= 3 * M) return;
  int c = i / M, p = i - c * M;
  int y = p / 92, x = p - y * 92;
  float a = alb[((size_t)c * 128 + 18 + y) * 128 + 18 + x];
  out[i] = a * rd[i] + (__expf(rs[i]) - 1.f);
}

// ---------------------------------------------------------------------------
// Host-side orchestration
// ---------------------------------------------------------------------------
struct WS { char* base; size_t off; };
static void* wsAlloc(WS& w, size_t bytes) {
  void* p = w.base + w.off;
  w.off = (w.off + bytes + 255) & ~(size_t)255;
  return p;
}

static void launch_conv(hipStream_t s, const float* w32, const float* b32,
                        const _Float16* X, _Float16* Y16, _Float16* wscr,
                        int Cin, int Hin, int Win, int Cout, int Hout, int Wout,
                        int ks, int stride, int pad, int relu)
{
  int cinkk = Cin * ks * ks;
  int Kpad  = cdiv(cinkk, BLK_K) * BLK_K;
  int Npad  = cdiv(Cout, BLK_N) * BLK_N;
  int total = Npad * Kpad;
  pack_weights<<<cdiv(total, 256), 256, 0, s>>>(w32, wscr, Cout, cinkk, Kpad, total);
  dim3 grid(cdiv(Hout * Wout, BLK_M), cdiv(Cout, BLK_N));
  conv_wmma_f16<<<grid, 256, 0, s>>>(X, wscr, b32, Y16,
                                     Cin, Hin, Win, Cout, Hout, Wout,
                                     ks, stride, pad, relu, Kpad);
}

// conv chain: nl layers, 5x5; valid=true -> VALID (shrink 4/layer), else SAME pad 2
static void run_chain(hipStream_t s, void* const* din, int pb, int nl,
                      int cin0, int mid, int clast, const _Float16* x0, int H0, bool valid,
                      _Float16* ping, _Float16* pong, _Float16* dst, _Float16* wscr)
{
  const _Float16* x = x0;
  int H = H0;
  for (int i = 0; i < nl; ++i) {
    int cin  = (i == 0) ? cin0 : mid;
    int cout = (i == nl - 1) ? clast : mid;
    int Ho   = valid ? (H - 4) : H;
    int pad  = valid ? 0 : 2;
    _Float16* y = (i == nl - 1) ? dst : ((i & 1) ? pong : ping);
    launch_conv(s, (const float*)din[pb + 2 * i], (const float*)din[pb + 2 * i + 1],
                x, y, wscr, cin, H, H, cout, Ho, Ho, 5, 1, pad, i < nl - 1);
    x = y;
    H = Ho;
  }
}

// SimpleUNet at 92x92: e1 -> e2(stride2) -> up2 -> u -> cat(u,e1) -> d -> o(1x1,51)
static void run_unet(hipStream_t s, void* const* din, int pb, const _Float16* x,
                     _Float16* e1, _Float16* e2, _Float16* up, _Float16* cat,
                     _Float16* dbuf, _Float16* o51, _Float16* wscr)
{
  const int M = 92 * 92;
  launch_conv(s, (const float*)din[pb + 0], (const float*)din[pb + 1],
              x, e1, wscr, 34, 92, 92, 64, 92, 92, 3, 1, 1, 1);
  launch_conv(s, (const float*)din[pb + 2], (const float*)din[pb + 3],
              e1, e2, wscr, 64, 92, 92, 128, 46, 46, 3, 2, 0, 1);
  up2<<<cdiv(128 * M, 256), 256, 0, s>>>(e2, up, 128, 46, 46);
  launch_conv(s, (const float*)din[pb + 4], (const float*)din[pb + 5],
              up, cat, wscr, 128, 92, 92, 64, 92, 92, 3, 1, 1, 1);
  (void)hipMemcpyAsync(cat + (size_t)64 * M, e1, (size_t)64 * M * sizeof(_Float16),
                       hipMemcpyDeviceToDevice, s);
  launch_conv(s, (const float*)din[pb + 6], (const float*)din[pb + 7],
              cat, dbuf, wscr, 128, 92, 92, 64, 92, 92, 3, 1, 1, 1);
  launch_conv(s, (const float*)din[pb + 8], (const float*)din[pb + 9],
              dbuf, o51, wscr, 64, 92, 92, 51, 92, 92, 1, 1, 0, 0);
}

extern "C" void kernel_launch(void* const* d_in, const int* in_sizes, int n_in,
                              void* d_out, int out_size, void* d_ws, size_t ws_size,
                              hipStream_t stream)
{
  (void)in_sizes; (void)n_in; (void)out_size; (void)ws_size;
  const int HW  = 128 * 128;
  const int M92 = 92 * 92;

  WS ws{ (char*)d_ws, 0 };
  auto a16 = [&](size_t n) { return (_Float16*)wsAlloc(ws, n * sizeof(_Float16)); };
  auto a32 = [&](size_t n) { return (float*)wsAlloc(ws, n * sizeof(float)); };

  _Float16* wscr  = a16((size_t)448 * 2560);         // max Npad*Kpad weight panel
  _Float16* ping  = a16((size_t)100 * 124 * 124);
  _Float16* pong  = a16((size_t)100 * 124 * 124);
  _Float16* smA   = a16((size_t)441 * M92);
  _Float16* smB   = a16((size_t)441 * M92);
  _Float16* kdi16 = a16((size_t)34 * HW);
  _Float16* ksi16 = a16((size_t)34 * HW);
  _Float16* pdi16 = a16((size_t)17 * HW);
  _Float16* psi16 = a16((size_t)17 * HW);
  _Float16* gkd   = a16((size_t)441 * M92);
  _Float16* gks   = a16((size_t)441 * M92);
  _Float16* pkd   = a16((size_t)441 * M92);
  _Float16* pks   = a16((size_t)441 * M92);
  _Float16* fd16  = a16((size_t)31 * M92);
  _Float16* fs16  = a16((size_t)31 * M92);
  _Float16* disin = a16((size_t)34 * M92);
  _Float16* e1    = a16((size_t)64 * M92);
  _Float16* e2b   = a16((size_t)128 * 46 * 46);
  _Float16* upb   = a16((size_t)128 * M92);
  _Float16* catb  = a16((size_t)128 * M92);
  _Float16* dbuf  = a16((size_t)64 * M92);
  _Float16* o51   = a16((size_t)51 * M92);
  _Float16* rdin  = a16((size_t)100 * M92);
  _Float16* rsin  = a16((size_t)100 * M92);
  _Float16* wdr   = a16((size_t)M92);
  _Float16* wsr   = a16((size_t)M92);
  float* grd = a32((size_t)3 * M92);
  float* grs = a32((size_t)3 * M92);
  float* prd = a32((size_t)3 * M92);
  float* prs = a32((size_t)3 * M92);
  float* rrd = a32((size_t)3 * M92);
  float* rrs = a32((size_t)3 * M92);

  const float* IN_KDI = (const float*)d_in[0];
  const float* IN_KSI = (const float*)d_in[1];
  const float* IN_PD  = (const float*)d_in[2];
  const float* IN_PS  = (const float*)d_in[3];
  const float* BUFD   = (const float*)d_in[4];
  const float* BUFS   = (const float*)d_in[5];
  const float* ALB    = (const float*)d_in[6];

  // ---- input casts & concats (f16) ----
  cast_crop<<<cdiv(34 * HW, 256), 256, 0, stream>>>(IN_KDI, kdi16, 34, 128, 128, 0, 128, 128);
  cast_crop<<<cdiv(34 * HW, 256), 256, 0, stream>>>(IN_KSI, ksi16, 34, 128, 128, 0, 128, 128);
  cast_crop<<<cdiv(10 * HW, 256), 256, 0, stream>>>(IN_KDI, pdi16, 10, 128, 128, 0, 128, 128);
  cast_crop<<<cdiv(7 * HW, 256), 256, 0, stream>>>(IN_PD, pdi16 + (size_t)10 * HW, 7, 128, 128, 0, 128, 128);
  cast_crop<<<cdiv(10 * HW, 256), 256, 0, stream>>>(IN_KSI, psi16, 10, 128, 128, 0, 128, 128);
  cast_crop<<<cdiv(7 * HW, 256), 256, 0, stream>>>(IN_PS, psi16 + (size_t)10 * HW, 7, 128, 128, 0, 128, 128);

  // ---- 4 KPCN kernel-prediction chains (VALID 5x5 x9) ----
  run_chain(stream, d_in,  9, 9, 34, 100, 441, kdi16, 128, true, ping, pong, gkd, wscr);
  run_chain(stream, d_in, 27, 9, 34, 100, 441, ksi16, 128, true, ping, pong, gks, wscr);
  run_chain(stream, d_in, 45, 9, 17, 100, 441, pdi16, 128, true, ping, pong, pkd, wscr);
  run_chain(stream, d_in, 63, 9, 17, 100, 441, psi16, 128, true, ping, pong, pks, wscr);

  // ---- softmax + weighted filters (g_r_*, p_r_*) ----
  softmaxC<<<cdiv(M92, 128), 128, 0, stream>>>(gkd, smA, 441, M92);
  wfilter<<<cdiv(M92, 128), 128, 0, stream>>>(BUFD, smA, grd);
  softmaxC<<<cdiv(M92, 128), 128, 0, stream>>>(gks, smA, 441, M92);
  wfilter<<<cdiv(M92, 128), 128, 0, stream>>>(BUFS, smA, grs);
  softmaxC<<<cdiv(M92, 128), 128, 0, stream>>>(pkd, smA, 441, M92);
  wfilter<<<cdiv(M92, 128), 128, 0, stream>>>(BUFD, smA, prd);
  softmaxC<<<cdiv(M92, 128), 128, 0, stream>>>(pks, smA, 441, M92);
  wfilter<<<cdiv(M92, 128), 128, 0, stream>>>(BUFS, smA, prs);

  // ---- feature crops (31ch, crop 128->92 @18) ----
  cast_crop<<<cdiv(24 * M92, 256), 256, 0, stream>>>(IN_KDI + (size_t)10 * HW, fd16, 24, 128, 128, 18, 92, 92);
  cast_crop<<<cdiv(7 * M92, 256), 256, 0, stream>>>(IN_PD, fd16 + (size_t)24 * M92, 7, 128, 128, 18, 92, 92);
  cast_crop<<<cdiv(24 * M92, 256), 256, 0, stream>>>(IN_KSI + (size_t)10 * HW, fs16, 24, 128, 128, 18, 92, 92);
  cast_crop<<<cdiv(7 * M92, 256), 256, 0, stream>>>(IN_PS, fs16 + (size_t)24 * M92, 7, 128, 128, 18, 92, 92);

  // ---- discriminator (diffuse): G then P, shared Dd params @101 ----
  cast_crop<<<cdiv(3 * M92, 256), 256, 0, stream>>>(grd, disin, 3, 92, 92, 0, 92, 92);
  (void)hipMemcpyAsync(disin + (size_t)3 * M92, fd16, (size_t)31 * M92 * sizeof(_Float16),
                       hipMemcpyDeviceToDevice, stream);
  run_unet(stream, d_in, 101, disin, e1, e2b, upb, catb, dbuf, o51, wscr);
  sigmoid_slice<<<cdiv(50 * M92, 256), 256, 0, stream>>>(o51, rdin, 50, M92);

  cast_crop<<<cdiv(3 * M92, 256), 256, 0, stream>>>(prd, disin, 3, 92, 92, 0, 92, 92);
  run_unet(stream, d_in, 101, disin, e1, e2b, upb, catb, dbuf, o51, wscr);
  sigmoid_slice<<<cdiv(50 * M92, 256), 256, 0, stream>>>(o51, rdin + (size_t)50 * M92, 50, M92);

  // ---- discriminator (specular): G then P, shared Ds params @111 ----
  cast_crop<<<cdiv(3 * M92, 256), 256, 0, stream>>>(grs, disin, 3, 92, 92, 0, 92, 92);
  (void)hipMemcpyAsync(disin + (size_t)3 * M92, fs16, (size_t)31 * M92 * sizeof(_Float16),
                       hipMemcpyDeviceToDevice, stream);
  run_unet(stream, d_in, 111, disin, e1, e2b, upb, catb, dbuf, o51, wscr);
  sigmoid_slice<<<cdiv(50 * M92, 256), 256, 0, stream>>>(o51, rsin, 50, M92);

  cast_crop<<<cdiv(3 * M92, 256), 256, 0, stream>>>(prs, disin, 3, 92, 92, 0, 92, 92);
  run_unet(stream, d_in, 111, disin, e1, e2b, upb, catb, dbuf, o51, wscr);
  sigmoid_slice<<<cdiv(50 * M92, 256), 256, 0, stream>>>(o51, rsin + (size_t)50 * M92, 50, M92);

  // ---- blend weight chains (SAME 5x5 x5, 100->...->1) ----
  run_chain(stream, d_in, 81, 5, 100, 100, 1, rdin, 92, false, ping, pong, wdr, wscr);
  run_chain(stream, d_in, 91, 5, 100, 100, 1, rsin, 92, false, ping, pong, wsr, wscr);

  // ---- kernel interpolation + final filters ----
  blendk<<<cdiv(441 * M92, 256), 256, 0, stream>>>(gkd, pkd, wdr, smA, 441, M92);
  softmaxC<<<cdiv(M92, 128), 128, 0, stream>>>(smA, smB, 441, M92);
  wfilter<<<cdiv(M92, 128), 128, 0, stream>>>(BUFD, smB, rrd);

  blendk<<<cdiv(441 * M92, 256), 256, 0, stream>>>(gks, pks, wsr, smA, 441, M92);
  softmaxC<<<cdiv(M92, 128), 128, 0, stream>>>(smA, smB, 441, M92);
  wfilter<<<cdiv(M92, 128), 128, 0, stream>>>(BUFS, smB, rrs);

  // ---- final composite ----
  final_out<<<cdiv(3 * M92, 256), 256, 0, stream>>>(ALB, rrd, rrs, (float*)d_out);
}
